// ContrastivePNAConv_40381282517157
// MI455X (gfx1250) — compile-verified, matrix operand order checked by reading, not run on hardware
//
#include <hip/hip_runtime.h>

typedef __attribute__((ext_vector_type(8)))  float  v8f;
typedef __attribute__((ext_vector_type(16))) __bf16 v16bf;
typedef __attribute__((ext_vector_type(8)))  __bf16 v8bf;

#define NB      8
#define NNODES  4096
#define DEG     16
#define UNITS   128
#define DEPTH   3
#define EDGES   (NNODES * DEG)
#define LOG10_17 1.2304489213782739f   // log(deg+1)/log(10), deg==16 for every node

// ---------------------------------------------------------------------------
// Prep: fold the 9-block PNA concat into Ws_eff / Wm_eff, fold BN into affine,
// and swizzle the effective weights into the CDNA5 16-bit B-fragment layout
// so the GEMM kernel does contiguous 32B-per-lane fragment loads.
// B-fragment mapping (16-bit, 32x16 K-chunk): lane = (n&15) | (((kk>>3)&1)<<4),
// elem = (kk&7) + ((kk&16)>>1)   (kk = k within 32-chunk)
// ---------------------------------------------------------------------------
__global__ void pna_prep(const float* __restrict__ W,
                         const float* __restrict__ gamma,
                         const float* __restrict__ beta,
                         const float* __restrict__ mmean,
                         const float* __restrict__ mvar,
                         __bf16* __restrict__ wsfrag,
                         __bf16* __restrict__ wmfrag,
                         float* __restrict__ aP,
                         float* __restrict__ b2P) {
    int tid = blockIdx.x * blockDim.x + threadIdx.x;
    if (tid >= DEPTH * UNITS * UNITS) return;
    int l = tid >> 14;
    int rem = tid & 16383;
    int k = rem >> 7;        // contraction index 0..127
    int j = rem & 127;       // output channel 0..127

    const float c = LOG10_17;
    const size_t base = ((size_t)l * 9 * UNITS) * UNITS + (size_t)k * UNITS + j;
    #define WBLK(blk) W[base + (size_t)(blk) * UNITS * UNITS]
    float ws = (WBLK(0) + c * (WBLK(1) + WBLK(2))) * 0.0625f
             +  WBLK(6) + c * (WBLK(7) + WBLK(8));
    float wm =  WBLK(3) + c * (WBLK(4) + WBLK(5));
    #undef WBLK

    int cch  = k >> 5;
    int kk   = k & 31;
    int t    = j >> 4;
    int lane = (j & 15) | (((kk >> 3) & 1) << 4);
    int e    = (kk & 7) + ((kk & 16) >> 1);
    size_t idx = ((((size_t)l * 4 + cch) * 8 + t) * 32 + lane) * 16 + e;
    wsfrag[idx] = (__bf16)ws;
    wmfrag[idx] = (__bf16)wm;

    if (k == 0) {
        int o = l * UNITS + j;
        float av = gamma[o] * rsqrtf(mvar[o] + 1e-3f);
        aP[o]  = av;
        b2P[o] = beta[o] - mmean[o] * av;
    }
}

// ---------------------------------------------------------------------------
// Aggregation: one wave per (batch, node). Edges of node n are contiguous
// [n*16, n*16+16). Each lane owns 4 channels (float4). sum + max in f32,
// stored as bf16 WMMA A-operands.
// ---------------------------------------------------------------------------
__global__ void pna_agg(const float* __restrict__ x,
                        const int* __restrict__ ei,
                        __bf16* __restrict__ ssum,
                        __bf16* __restrict__ smax) {
    int wave = (blockIdx.x * blockDim.x + threadIdx.x) >> 5;
    int lane = threadIdx.x & 31;
    int b = wave >> 12;          // 4096 waves per batch
    int node = wave & (NNODES - 1);

    const int* e = ei + ((size_t)b * EDGES + (size_t)node * DEG) * 2;
    const float* xb = x + (size_t)b * NNODES * UNITS;

    float sx = 0.f, sy = 0.f, sz = 0.f, sw = 0.f;
    float mx = -3.402823466e38f, my = mx, mz = mx, mw = mx;
    #pragma unroll
    for (int i = 0; i < DEG; ++i) {
        int d = e[i * 2 + 1];
        const float4 v = *(const float4*)(xb + (size_t)d * UNITS + lane * 4);
        sx += v.x; sy += v.y; sz += v.z; sw += v.w;
        mx = fmaxf(mx, v.x); my = fmaxf(my, v.y);
        mz = fmaxf(mz, v.z); mw = fmaxf(mw, v.w);
    }
    size_t o = ((size_t)b * NNODES + node) * UNITS + lane * 4;
    ssum[o + 0] = (__bf16)sx; ssum[o + 1] = (__bf16)sy;
    ssum[o + 2] = (__bf16)sz; ssum[o + 3] = (__bf16)sw;
    smax[o + 0] = (__bf16)mx; smax[o + 1] = (__bf16)my;
    smax[o + 2] = (__bf16)mz; smax[o + 3] = (__bf16)mw;
}

// ---------------------------------------------------------------------------
// GEMM: one wave per 16x16 output tile. z = S@Ws + M@Wm, K=128 in 4 chunks of
// 32 -> 8 v_wmma_f32_16x16x32_bf16 per wave. Fused epilogue: +bias, ReLU,
// folded-BN affine, f32 store.
// A layout (16-bit 16x32): lane l -> row = tile+ (l&15); h = l>>4 selects
// K-halves {h*8..h*8+7} U {16+h*8..16+h*8+7}  == two contiguous 16B loads.
// ---------------------------------------------------------------------------
__global__ void pna_gemm(const __bf16* __restrict__ ssum,
                         const __bf16* __restrict__ smax,
                         const __bf16* __restrict__ wsfrag,
                         const __bf16* __restrict__ wmfrag,
                         const float* __restrict__ bias,
                         const float* __restrict__ aP,
                         const float* __restrict__ b2P,
                         float* __restrict__ xout,
                         int layer) {
    int wid  = (blockIdx.x * blockDim.x + threadIdx.x) >> 5;
    int lane = threadIdx.x & 31;
    int b  = wid >> 11;          // 2048 tiles per batch
    int rem = wid & 2047;
    int rt = rem >> 3;           // row tile 0..255
    int ct = rem & 7;            // col tile 0..7
    int h  = lane >> 4;
    int lm = lane & 15;

    const size_t bo = (size_t)b * NNODES * UNITS;
    const size_t arow = bo + (size_t)(rt * 16 + lm) * UNITS + h * 8;
    const size_t fbase = (((size_t)layer * 4) * 8 + ct) * 32 + lane;

    v8f acc = {};
    #pragma unroll
    for (int c = 0; c < 4; ++c) {
        size_t ao = arow + c * 32;
        v8bf slo = *(const v8bf*)(ssum + ao);
        v8bf shi = *(const v8bf*)(ssum + ao + 16);
        v16bf aS = __builtin_shufflevector(slo, shi,
            0,1,2,3,4,5,6,7,8,9,10,11,12,13,14,15);
        v16bf bS = *(const v16bf*)(wsfrag + (fbase + (size_t)c * 8 * 32) * 16);
        acc = __builtin_amdgcn_wmma_f32_16x16x32_bf16(
            false, aS, false, bS, (short)0, acc, false, false);

        v8bf mlo = *(const v8bf*)(smax + ao);
        v8bf mhi = *(const v8bf*)(smax + ao + 16);
        v16bf aM = __builtin_shufflevector(mlo, mhi,
            0,1,2,3,4,5,6,7,8,9,10,11,12,13,14,15);
        v16bf bM = *(const v16bf*)(wmfrag + (fbase + (size_t)c * 8 * 32) * 16);
        acc = __builtin_amdgcn_wmma_f32_16x16x32_bf16(
            false, aM, false, bM, (short)0, acc, false, false);
    }

    // C/D layout: VGPR r, lanes<16 -> M=r, lanes>=16 -> M=8+r; N = lane&15
    int col  = ct * 16 + lm;
    int row0 = rt * 16 + 8 * h;
    float av  = aP[col];
    float b2v = b2P[col];
    float biv = bias[col];
    #pragma unroll
    for (int r = 0; r < 8; ++r) {
        float z = fmaxf(acc[r] + biv, 0.f);           // relu(feats@W + b)
        xout[bo + (size_t)(row0 + r) * UNITS + col] = z * av + b2v;  // folded BN
    }
}

// ---------------------------------------------------------------------------
// Graph readout: mean over nodes. One block per batch, one thread per channel.
// ---------------------------------------------------------------------------
__global__ void pna_mean(const float* __restrict__ x, float* __restrict__ g) {
    int b = blockIdx.x;
    int j = threadIdx.x;
    const float* xb = x + (size_t)b * NNODES * UNITS;
    float s = 0.f;
    for (int n = 0; n < NNODES; ++n) s += xb[(size_t)n * UNITS + j];
    g[b * UNITS + j] = s * (1.0f / NNODES);
}

// ---------------------------------------------------------------------------
// Projection MLP: [8,128]@[128,128]+relu -> [8,128]@[128,64]+relu. Tiny.
// ---------------------------------------------------------------------------
__global__ void pna_mlp(const float* __restrict__ g,
                        const float* __restrict__ Wp1, const float* __restrict__ bp1,
                        const float* __restrict__ Wp2, const float* __restrict__ bp2,
                        float* __restrict__ out) {
    __shared__ float h1[NB][UNITS];
    int t = threadIdx.x;
    for (int b = 0; b < NB; ++b) {
        float s = bp1[t];
        for (int k = 0; k < UNITS; ++k) s += g[b * UNITS + k] * Wp1[k * UNITS + t];
        h1[b][t] = fmaxf(s, 0.f);
    }
    __syncthreads();
    if (t < UNITS / 2) {
        for (int b = 0; b < NB; ++b) {
            float s = bp2[t];
            for (int k = 0; k < UNITS; ++k) s += h1[b][k] * Wp2[k * (UNITS / 2) + t];
            out[b * (UNITS / 2) + t] = fmaxf(s, 0.f);
        }
    }
}

// ---------------------------------------------------------------------------
extern "C" void kernel_launch(void* const* d_in, const int* in_sizes, int n_in,
                              void* d_out, int out_size, void* d_ws, size_t ws_size,
                              hipStream_t stream) {
    const float* nodeAttr = (const float*)d_in[0];
    const int*   edges    = (const int*)d_in[1];
    const float* W        = (const float*)d_in[2];
    const float* bvec     = (const float*)d_in[3];
    const float* gamma    = (const float*)d_in[4];
    const float* beta     = (const float*)d_in[5];
    const float* mmean    = (const float*)d_in[6];
    const float* mvar     = (const float*)d_in[7];
    const float* Wp1      = (const float*)d_in[8];
    const float* bp1      = (const float*)d_in[9];
    const float* Wp2      = (const float*)d_in[10];
    const float* bp2      = (const float*)d_in[11];
    float* out = (float*)d_out;

    char* ws = (char*)d_ws;
    const size_t FRAG_BYTES = (size_t)DEPTH * 4 * 8 * 32 * 16 * 2;   // 98304
    size_t off = 0;
    __bf16* wsfrag = (__bf16*)(ws + off); off += FRAG_BYTES;
    __bf16* wmfrag = (__bf16*)(ws + off); off += FRAG_BYTES;
    float*  aP     = (float*)(ws + off);  off += DEPTH * UNITS * sizeof(float);
    float*  b2P    = (float*)(ws + off);  off += DEPTH * UNITS * sizeof(float);
    off = (off + 255) & ~(size_t)255;
    __bf16* ssum   = (__bf16*)(ws + off); off += (size_t)NB * NNODES * UNITS * 2;
    __bf16* smax   = (__bf16*)(ws + off); off += (size_t)NB * NNODES * UNITS * 2;
    float*  x0     = (float*)(ws + off);  off += (size_t)NB * NNODES * UNITS * 4;
    float*  x1     = (float*)(ws + off);  off += (size_t)NB * NNODES * UNITS * 4;
    float*  gbuf   = (float*)(ws + off);  off += (size_t)NB * UNITS * 4;

    // 1) fold weights + BN, swizzle WMMA fragments
    pna_prep<<<(DEPTH * UNITS * UNITS + 255) / 256, 256, 0, stream>>>(
        W, gamma, beta, mmean, mvar, wsfrag, wmfrag, aP, b2P);

    // 2) three PNA layers: gather-reduce then WMMA GEMM with fused epilogue
    const float* xin[DEPTH]  = {nodeAttr, x0, x1};
    float*       xout[DEPTH] = {x0, x1, x0};
    for (int l = 0; l < DEPTH; ++l) {
        pna_agg<<<(NB * NNODES) / 8, 256, 0, stream>>>(xin[l], edges, ssum, smax);
        pna_gemm<<<(NB * NNODES / 16 * 8) / 8, 256, 0, stream>>>(
            ssum, smax, wsfrag, wmfrag,
            bvec + l * UNITS, aP + l * UNITS, b2P + l * UNITS,
            xout[l], l);
    }

    // 3) readout mean + projection MLP
    pna_mean<<<NB, UNITS, 0, stream>>>(x0, gbuf);
    pna_mlp<<<1, UNITS, 0, stream>>>(gbuf, Wp1, bp1, Wp2, bp2, out);
}
